// Net_80573586473382
// MI455X (gfx1250) — matrix operand order repaired and checked by reading, not compiled, not run on hardware
//
#include <hip/hip_runtime.h>
#include <hip/hip_bf16.h>
#include <math.h>

#define EPSV 1e-5f

typedef __bf16 bf16;
typedef __attribute__((ext_vector_type(16))) bf16 v16bf;
typedef __attribute__((ext_vector_type(8)))  float v8f;

union Frag  { v16bf v; uint4 u[2]; };
union U4H8  { uint4 u; bf16 h[8]; };

// ---------------------------------------------------------------------------
// Implicit-GEMM conv / GEMM kernel. bf16 in (X, W, res), f32 accumulate,
// bf16 or f32 out. Tile 128(M) x 128(N) x 32(K); 256 threads = 8 waves,
// each wave owns 32x64 = 2x4 WMMA accumulators.
// Double-buffered LDS pipeline:
//   - A (weights) tile staged global->LDS with GLOBAL_LOAD_ASYNC_TO_LDS_B128
//     (ASYNCcnt) when K % 32 == 0 (all convs except stem K=147, f9 K=2051);
//   - B tile: 1x1/s1/p0 fast path = uint4 (8 x bf16) loads along contiguous
//     pixels; generic im2col path uses incremental k decomposition;
//   - staging of tile t+1 is issued BEFORE the WMMAs of tile t, so async
//     copies and B global loads overlap matrix math; one s_wait_asynccnt 0
//     + barrier per iteration publishes the next buffer.
// ---------------------------------------------------------------------------
__global__ __launch_bounds__(256) void conv_wmma(
    const bf16* __restrict__ X, const bf16* __restrict__ W,
    void* __restrict__ Yv,
    const float* __restrict__ scale, const float* __restrict__ bias,
    const bf16* __restrict__ res, int do_relu, int out_f32, float alpha,
    int Nb, int IC, int IH, int IW, int OC, int OH, int OW,
    int KH, int KW, int stride, int pad, int fast1x1)
{
    const int K   = IC * KH * KW;
    const int OHW = OH * OW;
    const int P   = Nb * OHW;
    const int KHW = KH * KW;

    __shared__ __align__(16) bf16 As[2][128 * 40];  // [m][k], row stride 40
    __shared__ __align__(16) bf16 Bs[2][128 * 40];  // [n][k] (transposed)

    const int tid  = threadIdx.x;
    const int n0   = blockIdx.x * 128;   // pixel base
    const int m0   = blockIdx.y * 128;   // out-channel base
    const int lane = tid & 31;
    const int wv   = tid >> 5;
    const int wM   = wv & 3;
    const int wN   = wv >> 2;
    const int hf   = lane >> 4;
    const int l16  = lane & 15;

    v8f acc[2][4];
    const v8f vzero = {0.f,0.f,0.f,0.f,0.f,0.f,0.f,0.f};
#pragma unroll
    for (int i = 0; i < 2; ++i)
#pragma unroll
        for (int j = 0; j < 4; ++j) acc[i][j] = vzero;

    // ---- per-thread staging precompute (hoisted divisions) ----
    const int ldRow = tid >> 1;
    const int kseg  = (tid & 1) * 16;
    const bool fastA = ((K & 31) == 0);

    // generic B: pixel of this thread's row
    const int np = n0 + ldRow;
    int bbg = 0, oyg = 0, oxg = 0;
    const bool npv = (np < P);
    if (npv && !fast1x1) {
        bbg = np / OHW; int p = np - bbg * OHW; oyg = p / OW; oxg = p - oyg * OW;
    }
    // fast-1x1 B: thread covers k = tid&31, pixels ng*16 .. ng*16+15 (2 x uint4)
    const int kloc = tid & 31;
    const int ng   = tid >> 5;
    int fb[2], fp[2]; bool fv[2];
    if (fast1x1) {
#pragma unroll
        for (int j = 0; j < 2; ++j) {
            int nb = n0 + ng * 16 + j * 8;
            fv[j] = (nb < P);
            if (fv[j]) { fb[j] = nb / OHW; fp[j] = nb - fb[j] * OHW; }
            else       { fb[j] = 0; fp[j] = 0; }
        }
    }

    auto stageA = [&](int k0, bf16* Asb) {
        const int m = m0 + ldRow;
        if (fastA) {
            if (m < OC) {
                size_t   g = (size_t)(W + (size_t)m * K + (k0 + kseg));
                unsigned l = (unsigned)(size_t)&Asb[ldRow * 40 + kseg];
                asm volatile("global_load_async_to_lds_b128 %0, %1, off"
                             :: "v"(l), "v"((unsigned long long)g) : "memory");
                asm volatile("global_load_async_to_lds_b128 %0, %1, off"
                             :: "v"((unsigned)(l + 16)),
                                "v"((unsigned long long)(g + 16)) : "memory");
            } // m>=OC rows left stale; epilogue masks oc>=OC
        } else {
#pragma unroll
            for (int kk = 0; kk < 16; ++kk) {
                const int k = k0 + kseg + kk;
                bf16 v = (bf16)0.f;
                if (m < OC && k < K) v = W[(size_t)m * K + k];
                Asb[ldRow * 40 + kseg + kk] = v;
            }
        }
    };

    auto stageB = [&](int k0, bf16* Bsb) {
        if (fast1x1) {
            const int k = k0 + kloc;
            const bool kv = (k < K);
#pragma unroll
            for (int j = 0; j < 2; ++j) {
                const int nloc = ng * 16 + j * 8;
                if (kv && fv[j]) {
                    U4H8 q;
                    q.u = *(const uint4*)(X + ((size_t)fb[j] * IC + k) * OHW + fp[j]);
#pragma unroll
                    for (int e = 0; e < 8; ++e)
                        Bsb[(nloc + e) * 40 + kloc] = q.h[e];
                } else {
#pragma unroll
                    for (int e = 0; e < 8; ++e)
                        Bsb[(nloc + e) * 40 + kloc] = (bf16)0.f;
                }
            }
        } else {
            int kb = k0 + kseg;
            int c  = kb / KHW; int r = kb - c * KHW;
            int ky = r / KW;   int kx = r - ky * KW;
#pragma unroll
            for (int kk = 0; kk < 16; ++kk) {
                const int k = kb + kk;
                bf16 v = (bf16)0.f;
                if (npv && k < K) {
                    const int iy = oyg * stride - pad + ky;
                    const int ix = oxg * stride - pad + kx;
                    if (iy >= 0 && iy < IH && ix >= 0 && ix < IW)
                        v = X[(((size_t)bbg * IC + c) * IH + iy) * IW + ix];
                }
                Bsb[ldRow * 40 + kseg + kk] = v;
                if (++kx == KW) { kx = 0; if (++ky == KH) { ky = 0; ++c; } }
            }
        }
    };

    // ---- software pipeline over K tiles ----
    const int nT = (K + 31) / 32;
    stageA(0, As[0]);
    stageB(0, Bs[0]);
    asm volatile("s_wait_asynccnt 0x0" ::: "memory");
    __syncthreads();

    for (int t = 0; t < nT; ++t) {
        const int cur = t & 1;
        bf16* curA = As[cur];
        bf16* curB = Bs[cur];
        if (t + 1 < nT) {                 // issue next tile's staging early
            stageA((t + 1) * 32, As[cur ^ 1]);
            stageB((t + 1) * 32, Bs[cur ^ 1]);
        }

        v16bf af[2], bfr[4];
#pragma unroll
        for (int mi = 0; mi < 2; ++mi) {
            const int rr = wM * 32 + mi * 16 + l16;
            Frag f;
            f.u[0] = *(const uint4*)&curA[rr * 40 + 8 * hf];
            f.u[1] = *(const uint4*)&curA[rr * 40 + 16 + 8 * hf];
            af[mi] = f.v;
        }
#pragma unroll
        for (int ni = 0; ni < 4; ++ni) {
            const int rr = wN * 64 + ni * 16 + l16;
            Frag f;
            f.u[0] = *(const uint4*)&curB[rr * 40 + 8 * hf];
            f.u[1] = *(const uint4*)&curB[rr * 40 + 16 + 8 * hf];
            bfr[ni] = f.v;
        }
#pragma unroll
        for (int mi = 0; mi < 2; ++mi)
#pragma unroll
            for (int ni = 0; ni < 4; ++ni)
                acc[mi][ni] = __builtin_amdgcn_wmma_f32_16x16x32_bf16(
                    false, af[mi], false, bfr[ni], (short)0, acc[mi][ni],
                    false, false);

        asm volatile("s_wait_asynccnt 0x0" ::: "memory");
        __syncthreads();
    }

    // ================= epilogue ===========================================
#pragma unroll
    for (int mi = 0; mi < 2; ++mi) {
#pragma unroll
        for (int ni = 0; ni < 4; ++ni) {
            const int n = n0 + wN * 64 + ni * 16 + l16;
            if (n >= P) continue;
            const int b = n / OHW;
            const int p = n - b * OHW;
#pragma unroll
            for (int r = 0; r < 8; ++r) {
                const int oc = m0 + wM * 32 + mi * 16 + r + 8 * hf;
                if (oc >= OC) continue;
                float v = acc[mi][ni][r] * alpha;
                if (scale) v = v * scale[oc] + bias[oc];
                const size_t idx = ((size_t)b * OC + oc) * OHW + p;
                if (res) v += (float)res[idx];
                if (do_relu) v = fmaxf(v, 0.f);
                if (out_f32) ((float*)Yv)[idx] = v;
                else         ((bf16*)Yv)[idx] = (bf16)v;
            }
        }
    }
}

// ---------------------------------------------------------------------------
// Helper kernels (bf16 activations, f32 accumulation / outputs)
// ---------------------------------------------------------------------------
__global__ void k_f2b(const float* __restrict__ in, bf16* __restrict__ out, long n)
{
    long t = (long)blockIdx.x * blockDim.x + threadIdx.x;
    if (t < n) out[t] = (bf16)in[t];
}

__global__ void k_maxpool(const bf16* __restrict__ in, bf16* __restrict__ out,
                          int NC, int IH, int IW, int OH, int OW)
{
    int t = blockIdx.x * blockDim.x + threadIdx.x;
    int total = NC * OH * OW;
    if (t >= total) return;
    int ox = t % OW; int tmp = t / OW;
    int oyy = tmp % OH; int nc = tmp / OH;
    float m = -3.0e38f;
    for (int dy = 0; dy < 3; ++dy) {
        int iy = oyy * 2 - 1 + dy;
        if (iy < 0 || iy >= IH) continue;
        for (int dx = 0; dx < 3; ++dx) {
            int ix = ox * 2 - 1 + dx;
            if (ix < 0 || ix >= IW) continue;
            m = fmaxf(m, (float)in[((size_t)nc * IH + iy) * IW + ix]);
        }
    }
    out[t] = (bf16)m;
}

__global__ void k_resize_bilinear(const bf16* __restrict__ in, bf16* __restrict__ out,
                                  int Nb, int C, int IH, int IW, int OH, int OW,
                                  int align, int coff, int Ctot)
{
    int t = blockIdx.x * blockDim.x + threadIdx.x;
    int total = Nb * C * OH * OW;
    if (t >= total) return;
    int ox = t % OW; int tmp = t / OW;
    int oyy = tmp % OH; tmp /= OH;
    int c = tmp % C; int n = tmp / C;
    float ys, xs;
    if (align) {
        ys = (OH > 1) ? oyy * (float)(IH - 1) / (float)(OH - 1) : 0.f;
        xs = (OW > 1) ? ox  * (float)(IW - 1) / (float)(OW - 1) : 0.f;
    } else {
        ys = fminf(fmaxf((oyy + 0.5f) * (float)IH / (float)OH - 0.5f, 0.f), (float)(IH - 1));
        xs = fminf(fmaxf((ox  + 0.5f) * (float)IW / (float)OW - 0.5f, 0.f), (float)(IW - 1));
    }
    int y0 = (int)floorf(ys); int y1 = min(y0 + 1, IH - 1); float wy = ys - (float)y0;
    int x0 = (int)floorf(xs); int x1 = min(x0 + 1, IW - 1); float wx = xs - (float)x0;
    const bf16* base = in + ((size_t)n * C + c) * IH * IW;
    float v = (1.f - wy) * ((1.f - wx) * (float)base[y0 * IW + x0] + wx * (float)base[y0 * IW + x1])
            +        wy  * ((1.f - wx) * (float)base[y1 * IW + x0] + wx * (float)base[y1 * IW + x1]);
    out[((size_t)n * Ctot + coff + c) * OH * OW + oyy * OW + ox] = (bf16)v;
}

__global__ void k_chan_l2norm(bf16* __restrict__ x, int Nb, int C, int HW, float eps)
{
    int t = blockIdx.x * blockDim.x + threadIdx.x;
    if (t >= Nb * HW) return;
    int p = t % HW; int n = t / HW;
    float s = 0.f;
    for (int c = 0; c < C; ++c) { float v = (float)x[((size_t)n * C + c) * HW + p]; s += v * v; }
    float inv = 1.f / (sqrtf(s) + eps);
    for (int c = 0; c < C; ++c) {
        size_t i = ((size_t)n * C + c) * HW + p;
        x[i] = (bf16)((float)x[i] * inv);
    }
}

__global__ void k_mean_hw(const bf16* __restrict__ x, float* __restrict__ out, int NC, int HW)
{
    int t = blockIdx.x * blockDim.x + threadIdx.x;
    if (t >= NC) return;
    float s = 0.f;
    for (int p = 0; p < HW; ++p) s += (float)x[(size_t)t * HW + p];
    out[t] = s / (float)HW;
}

__global__ void k_relumax_hw(const bf16* __restrict__ x, float* __restrict__ out, int NC, int HW)
{
    int t = blockIdx.x * blockDim.x + threadIdx.x;
    if (t >= NC) return;
    float m = 0.f;
    for (int p = 0; p < HW; ++p) m = fmaxf(m, fmaxf((float)x[(size_t)t * HW + p], 0.f));
    out[t] = m;
}

__global__ void k_build_nc(const bf16* __restrict__ cam, const float* __restrict__ mx,
                           const float* __restrict__ valid,
                           bf16* __restrict__ nc, float* __restrict__ orig,
                           int Nb, int HW)
{
    int t = blockIdx.x * blockDim.x + threadIdx.x;
    if (t >= Nb * HW) return;
    int p = t % HW; int n = t / HW;
    float vals[20];
    float mnorm = 0.f;
    for (int c = 0; c < 20; ++c) {
        float v = fmaxf((float)cam[((size_t)n * 20 + c) * HW + p], 0.f) / (mx[n * 20 + c] + EPSV);
        vals[c] = v;
        mnorm = fmaxf(mnorm, v);
    }
    float o = (1.f - mnorm) * valid[((size_t)n * 21) * HW + p];
    nc[((size_t)n * 21) * HW + p] = (bf16)o;
    orig[((size_t)n * 21) * HW + p] = o;
    for (int c = 0; c < 20; ++c) {
        float ov = vals[c] * valid[((size_t)n * 21 + c + 1) * HW + p];
        nc[((size_t)n * 21 + c + 1) * HW + p] = (bf16)ov;
        orig[((size_t)n * 21 + c + 1) * HW + p] = ov;
    }
}

__global__ void k_transpose(const bf16* __restrict__ in, bf16* __restrict__ out, int R, int C)
{
    int t = blockIdx.x * blockDim.x + threadIdx.x;
    if (t >= R * C) return;
    int c = t % C; int r = t / C;
    out[(size_t)c * R + r] = in[t];
}

__global__ void k_colsum_div(bf16* __restrict__ aff, int Pn)
{
    int q = blockIdx.x * blockDim.x + threadIdx.x;
    if (q >= Pn) return;
    float s = 0.f;
    for (int p = 0; p < Pn; ++p) s += (float)aff[(size_t)p * Pn + q];
    float inv = 1.f / (s + EPSV);
    for (int p = 0; p < Pn; ++p) {
        size_t i = (size_t)p * Pn + q;
        aff[i] = (bf16)((float)aff[i] * inv);
    }
}

__global__ void k_seeds(const float* __restrict__ cam1, const float* __restrict__ valid,
                        bf16* __restrict__ seeds, int Nb, int HW)
{
    int t = blockIdx.x * blockDim.x + threadIdx.x;
    if (t >= Nb * HW) return;
    int p = t % HW; int n = t / HW;
    float best = -3.0e38f; int bi = 0;
    for (int k = 0; k < 21; ++k) {
        float v = cam1[((size_t)n * 21 + k) * HW + p];
        if (k == 0) v *= 0.3f;
        if (v > best) { best = v; bi = k; }
    }
    for (int k = 0; k < 21; ++k)
        seeds[((size_t)n * 21 + k) * HW + p] =
            (k == bi) ? (bf16)valid[((size_t)n * 21 + k) * HW + p] : (bf16)0.f;
}

__global__ void k_hw_norm(const bf16* __restrict__ feat, float* __restrict__ fn,
                          int Nb, int C, int HW)
{
    int t = blockIdx.x * blockDim.x + threadIdx.x;
    if (t >= Nb * HW) return;
    int p = t % HW; int n = t / HW;
    float s = 0.f;
    for (int c = 0; c < C; ++c) { float v = (float)feat[((size_t)n * C + c) * HW + p]; s += v * v; }
    fn[t] = fmaxf(sqrtf(s), 1e-8f);
}

__global__ void k_row_norm(const bf16* __restrict__ proto, float* __restrict__ pn,
                           int NK, int C)
{
    int t = blockIdx.x * blockDim.x + threadIdx.x;
    if (t >= NK) return;
    float s = 0.f;
    for (int c = 0; c < C; ++c) { float v = (float)proto[(size_t)t * C + c]; s += v * v; }
    pn[t] = fmaxf(sqrtf(s), 1e-8f);
}

__global__ void k_cos(const float* __restrict__ dot, const float* __restrict__ fn,
                      const float* __restrict__ pn, float* __restrict__ out,
                      int Nb, int HW)
{
    int t = blockIdx.x * blockDim.x + threadIdx.x;
    if (t >= Nb * 21 * HW) return;
    int p = t % HW; int tmp = t / HW;
    int k = tmp % 21; int n = tmp / 21;
    out[t] = fmaxf(dot[t] / (fn[n * HW + p] * pn[n * 21 + k]), 0.f);
}

// ---------------------------------------------------------------------------
// Host-side orchestration
// ---------------------------------------------------------------------------
static inline void launch_conv(hipStream_t s, const bf16* X, const bf16* W, void* Y,
                               int Nb, int IC, int IH, int IW, int OC,
                               int KH, int KW, int stride, int pad,
                               const float* scale, const float* bias,
                               const bf16* res, int relu, int out_f32 = 0,
                               float alpha = 1.f)
{
    int OH = (IH + 2 * pad - KH) / stride + 1;
    int OW = (IW + 2 * pad - KW) / stride + 1;
    int P  = Nb * OH * OW;
    int fast1x1 = (KH == 1 && KW == 1 && stride == 1 && pad == 0 && ((OH * OW) % 8 == 0)) ? 1 : 0;
    dim3 g((P + 127) / 128, (OC + 127) / 128);
    conv_wmma<<<g, 256, 0, s>>>(X, W, Y, scale, bias, res, relu, out_f32, alpha,
                                Nb, IC, IH, IW, OC, OH, OW, KH, KW, stride, pad, fast1x1);
}

static inline void launch_gemm(hipStream_t s, const bf16* A, const bf16* B, void* C,
                               int M, int K, int N, int relu, int out_f32, float alpha)
{
    launch_conv(s, B, A, C, 1, K, 1, N, M, 1, 1, 1, 0,
                nullptr, nullptr, nullptr, relu, out_f32, alpha);
}

extern "C" void kernel_launch(void* const* d_in, const int* in_sizes, int n_in,
                              void* d_out_v, int out_size, void* d_ws, size_t ws_size,
                              hipStream_t stream)
{
    (void)in_sizes; (void)n_in; (void)out_size; (void)ws_size;
    const float* x_f   = (const float*)d_in[0];
    const float* valid = (const float*)d_in[1];
    float* dout = (float*)d_out_v;

    int pi = 2;
    auto nextp = [&]() -> const float* { return (const float*)d_in[pi++]; };

    // bump allocator over d_ws
    char* wsb = (char*)d_ws; size_t off = 0;
    auto allocBytes = [&](size_t nb) -> void* {
        void* p = (void*)(wsb + off);
        off += (nb + 255) & ~(size_t)255;
        return p;
    };
    auto allocF = [&](size_t n) -> float* { return (float*)allocBytes(n * sizeof(float)); };
    auto allocB = [&](size_t n) -> bf16*  { return (bf16*)allocBytes(n * sizeof(bf16)); };

    auto cvt = [&](const float* w, size_t n) -> const bf16* {
        bf16* p = allocB(n);
        k_f2b<<<(unsigned)((n + 255) / 256), 256, 0, stream>>>(w, p, (long)n);
        return p;
    };

    // ---- fetch + convert parameters (make_params insertion order) ----
    const bf16* w_conv1 = cvt(nextp(), (size_t)64 * 3 * 49);
    const float* s0 = nextp();
    const float* b0 = nextp();

    struct Blk { const bf16 *c1,*c2,*c3,*cd; const float *s1,*b1,*s2,*b2,*s3,*b3,*sd,*bd; };
    static const int NBLK[4]   = {3, 4, 6, 3};
    static const int PLANES[4] = {64, 128, 256, 512};
    Blk blks[4][6];
    int cin = 64;
    for (int li = 0; li < 4; ++li) {
        int pl = PLANES[li], cout = pl * 4;
        for (int bi = 0; bi < NBLK[li]; ++bi) {
            Blk& b = blks[li][bi];
            b.c1 = cvt(nextp(), (size_t)pl * cin);
            b.s1 = nextp(); b.b1 = nextp();
            b.c2 = cvt(nextp(), (size_t)pl * pl * 9);
            b.s2 = nextp(); b.b2 = nextp();
            b.c3 = cvt(nextp(), (size_t)cout * pl);
            b.s3 = nextp(); b.b3 = nextp();
            if (bi == 0) {
                b.cd = cvt(nextp(), (size_t)cout * cin);
                b.sd = nextp(); b.bd = nextp();
            } else { b.cd = nullptr; b.sd = nullptr; b.bd = nullptr; }
            cin = cout;
        }
    }
    static const int sIC[4] = {256, 512, 1024, 2048};
    static const int sOC[4] = {128, 128, 256, 256};
    const bf16* w_side[4];
    for (int i = 0; i < 4; ++i) w_side[i] = cvt(nextp(), (size_t)sOC[i] * sIC[i]);
    const bf16* w_cls = cvt(nextp(), (size_t)20 * 2048);
    const bf16* w_f9  = cvt(nextp(), (size_t)2048 * 2051);

    // ---- activations (bf16) ----
    const int Nb = 2;
    bf16* xb    = allocB((size_t)Nb * 3 * 512 * 512);
    bf16* h0    = allocB((size_t)Nb * 64 * 256 * 256);
    bf16* x0    = allocB((size_t)Nb * 64 * 128 * 128);
    bf16* x1    = allocB((size_t)Nb * 256 * 128 * 128);
    bf16* x2    = allocB((size_t)Nb * 512 * 64 * 64);
    bf16* x3    = allocB((size_t)Nb * 1024 * 32 * 32);
    bf16* x4    = allocB((size_t)Nb * 2048 * 32 * 32);
    bf16* pingA = allocB((size_t)Nb * 256 * 128 * 128);
    bf16* pingB = allocB((size_t)Nb * 256 * 128 * 128);
    bf16* t1    = allocB((size_t)Nb * 128 * 128 * 128);
    bf16* t2    = allocB((size_t)Nb * 64 * 128 * 128);
    bf16* scb   = allocB((size_t)Nb * 256 * 128 * 128);
    bf16* sbuf  = allocB((size_t)Nb * 128 * 128 * 128);
    bf16* hie   = allocB((size_t)Nb * 768 * 1024);
    bf16* cam   = allocB((size_t)Nb * 20 * 1024);
    float* mx   = allocF((size_t)Nb * 20);
    bf16* ncb   = allocB((size_t)Nb * 21 * 1024);
    bf16* fin   = allocB((size_t)Nb * 2051 * 1024);
    bf16* fbuf  = allocB((size_t)Nb * 2048 * 1024);
    bf16* fT    = allocB((size_t)1024 * 2048);
    bf16* aff   = allocB((size_t)1024 * 1024);
    bf16* ctmp  = allocB((size_t)21 * 1024);
    bf16* seeds = allocB((size_t)Nb * 21 * 1024);
    bf16* hieT  = allocB((size_t)1024 * 768);
    bf16* proto = allocB((size_t)Nb * 21 * 768);
    float* dotb = allocF((size_t)Nb * 21 * 1024);
    float* fn   = allocF((size_t)Nb * 1024);
    float* pn   = allocF((size_t)Nb * 21);

    k_f2b<<<(unsigned)(((size_t)Nb * 3 * 512 * 512 + 255) / 256), 256, 0, stream>>>(
        x_f, xb, (long)Nb * 3 * 512 * 512);

    // ---- stem ----
    launch_conv(stream, xb, w_conv1, h0, Nb, 3, 512, 512, 64, 7, 7, 2, 3, s0, b0, nullptr, 1);
    { int total = Nb * 64 * 128 * 128;
      k_maxpool<<<(total + 255) / 256, 256, 0, stream>>>(h0, x0, Nb * 64, 256, 256, 128, 128); }

    // ---- residual layers ----
    auto run_layer = [&](int li, const bf16* in, bf16* outF,
                         int cinL, int planes, int H, int W, int stride) {
        int cout = planes * 4;
        const bf16* cur = in;
        int curH = H, curW = W, curC = cinL;
        for (int bi = 0; bi < NBLK[li]; ++bi) {
            const Blk& B = blks[li][bi];
            int s  = (bi == 0) ? stride : 1;
            int H2 = (curH + 2 - 3) / s + 1;
            int W2 = (curW + 2 - 3) / s + 1;
            bf16* out = (bi == NBLK[li] - 1) ? outF : ((bi & 1) ? pingB : pingA);
            launch_conv(stream, cur, B.c1, t1, Nb, curC, curH, curW, planes,
                        1, 1, 1, 0, B.s1, B.b1, nullptr, 1);
            launch_conv(stream, t1, B.c2, t2, Nb, planes, curH, curW, planes,
                        3, 3, s, 1, B.s2, B.b2, nullptr, 1);
            const bf16* resp = cur;
            if (B.cd) {
                launch_conv(stream, cur, B.cd, scb, Nb, curC, curH, curW, cout,
                            1, 1, s, 0, B.sd, B.bd, nullptr, 0);
                resp = scb;
            }
            launch_conv(stream, t2, B.c3, out, Nb, planes, H2, W2, cout,
                        1, 1, 1, 0, B.s3, B.b3, resp, 1);
            cur = out; curC = cout; curH = H2; curW = W2;
        }
    };
    run_layer(0, x0, x1,   64,  64, 128, 128, 1);
    run_layer(1, x1, x2,  256, 128, 128, 128, 2);
    run_layer(2, x2, x3,  512, 256,  64,  64, 2);
    run_layer(3, x3, x4, 1024, 512,  32,  32, 1);

    // ---- side features -> normalized -> resized into hie_fea ----
    const bf16* sinp[4] = { x1, x2, x3, x4 };
    const int   sH[4]   = { 128, 64, 32, 32 };
    const int   coff[4] = { 0, 128, 256, 512 };
    for (int i = 0; i < 4; ++i) {
        int H = sH[i], hw = H * H;
        launch_conv(stream, sinp[i], w_side[i], sbuf, Nb, sIC[i], H, H, sOC[i],
                    1, 1, 1, 0, nullptr, nullptr, nullptr, 0);
        k_chan_l2norm<<<(Nb * hw + 255) / 256, 256, 0, stream>>>(sbuf, Nb, sOC[i], hw, EPSV);
        int total = Nb * sOC[i] * 32 * 32;
        k_resize_bilinear<<<(total + 255) / 256, 256, 0, stream>>>(
            sbuf, hie, Nb, sOC[i], H, H, 32, 32, 0, coff[i], 768);
    }

    // ---- classifier, score, nc / orignal_cam ----
    launch_conv(stream, x4, w_cls, cam, Nb, 2048, 32, 32, 20,
                1, 1, 1, 0, nullptr, nullptr, nullptr, 0);
    float* cam1o = dout + 40;
    float* cam2o = dout + 40 + (size_t)Nb * 21 * 1024;
    float* origo = dout + 40 + (size_t)2 * Nb * 21 * 1024;
    k_mean_hw<<<1, 64, 0, stream>>>(cam, dout, Nb * 20, 1024);
    k_relumax_hw<<<1, 64, 0, stream>>>(cam, mx, Nb * 20, 1024);
    k_build_nc<<<(Nb * 1024 + 255) / 256, 256, 0, stream>>>(cam, mx, valid, ncb, origo, Nb, 1024);

    // ---- f_in = concat(resize(x, 32, align=True), x4) ----
    { int total = Nb * 3 * 1024;
      k_resize_bilinear<<<(total + 255) / 256, 256, 0, stream>>>(
          xb, fin, Nb, 3, 512, 512, 32, 32, 1, 0, 2051); }
    for (int b = 0; b < Nb; ++b)
        hipMemcpyAsync(fin + ((size_t)b * 2051 + 3) * 1024,
                       x4 + (size_t)b * 2048 * 1024,
                       (size_t)2048 * 1024 * sizeof(bf16),
                       hipMemcpyDeviceToDevice, stream);

    // ---- PCM ----
    launch_conv(stream, fin, w_f9, fbuf, Nb, 2051, 32, 32, 2048,
                1, 1, 1, 0, nullptr, nullptr, nullptr, 0);
    k_chan_l2norm<<<(Nb * 1024 + 255) / 256, 256, 0, stream>>>(fbuf, Nb, 2048, 1024, EPSV);
    for (int b = 0; b < Nb; ++b) {
        const bf16* fb = fbuf + (size_t)b * 2048 * 1024;
        k_transpose<<<(2048 * 1024 + 255) / 256, 256, 0, stream>>>(fb, fT, 2048, 1024);
        launch_gemm(stream, fT, fb, aff, 1024, 2048, 1024, 1, 0, 1.f);
        k_colsum_div<<<(1024 + 255) / 256, 256, 0, stream>>>(aff, 1024);
        const bf16* ncp = ncb + (size_t)b * 21 * 1024;
        launch_gemm(stream, ncp, aff, ctmp, 21, 1024, 1024, 0, 0, 1.f);
        launch_gemm(stream, ctmp, aff, cam1o + (size_t)b * 21 * 1024, 21, 1024, 1024, 0, 1, 1.f);
    }

    // ---- prototype ----
    k_seeds<<<(Nb * 1024 + 255) / 256, 256, 0, stream>>>(cam1o, valid, seeds, Nb, 1024);
    for (int b = 0; b < Nb; ++b) {
        const bf16* hb = hie + (size_t)b * 768 * 1024;
        k_transpose<<<(768 * 1024 + 255) / 256, 256, 0, stream>>>(hb, hieT, 768, 1024);
        launch_gemm(stream, seeds + (size_t)b * 21 * 1024, hieT,
                    proto + (size_t)b * 21 * 768, 21, 1024, 768, 0, 0, 1.f / 1024.f);
        launch_gemm(stream, proto + (size_t)b * 21 * 768, hb,
                    dotb + (size_t)b * 21 * 1024, 21, 768, 1024, 0, 1, 1.f);
    }
    k_hw_norm<<<(Nb * 1024 + 255) / 256, 256, 0, stream>>>(hie, fn, Nb, 768, 1024);
    k_row_norm<<<1, 64, 0, stream>>>(proto, pn, Nb * 21, 768);
    k_cos<<<(Nb * 21 * 1024 + 255) / 256, 256, 0, stream>>>(dotb, fn, pn, cam2o, Nb, 1024);
}
